// MultiHeadAttention_7868380086545
// MI455X (gfx1250) — matrix operand order repaired and checked by reading, not compiled, not run on hardware
//
#include <hip/hip_runtime.h>
#include <hip/hip_bf16.h>

typedef __attribute__((ext_vector_type(16))) _Float16 v16h;
typedef __attribute__((ext_vector_type(8)))  _Float16 v8h;
typedef __attribute__((ext_vector_type(8)))  float    v8f;
typedef __attribute__((ext_vector_type(4)))  unsigned int u32x4;
typedef __attribute__((ext_vector_type(4)))  int      i32x4;
typedef __attribute__((ext_vector_type(8)))  int      i32x8;

#define D_MODEL  1024
#define NUM_HEAD 16
#define HEAD_DIM 64
#define BATCH    4
#define SEQ      2048
#define MROWS    (BATCH * SEQ)

// ---------------------------------------------------------------------------
// WMMA fragment loaders (wave32, 16x16x32 f16).
// A-matrix 16x32: lane group g = lane>>4 holds K = g*8+0..7 and 16+g*8+0..7.
// B-matrix 32x16: lane = column N, lane group g holds K = g*8+0..7 and 16+g*8+0..7.
// C/D 16x16 f32: element r <-> row r + 8*(lane>>4), column lane&15.
// ---------------------------------------------------------------------------
__device__ __forceinline__ v16h load_fragA(const _Float16* base, int stride, int lane) {
  const int g = lane >> 4;
  const int r = lane & 15;
  const _Float16* p = base + (size_t)r * stride + g * 8;
  v8h lo = *(const v8h*)(p);
  v8h hi = *(const v8h*)(p + 16);
  v16h f;
#pragma unroll
  for (int i = 0; i < 8; ++i) { f[i] = lo[i]; f[i + 8] = hi[i]; }
  return f;
}

__device__ __forceinline__ v16h load_fragB(const _Float16* base, int stride, int lane) {
  const int g = lane >> 4;
  const int n = lane & 15;
  const _Float16* p = base + (size_t)n * stride + g * 8;
  v8h lo = *(const v8h*)(p);
  v8h hi = *(const v8h*)(p + 16);
  v16h f;
#pragma unroll
  for (int i = 0; i < 8; ++i) { f[i] = lo[i]; f[i + 8] = hi[i]; }
  return f;
}

__device__ __forceinline__ v8f v8f_zero() {
  v8f z;
#pragma unroll
  for (int i = 0; i < 8; ++i) z[i] = 0.0f;
  return z;
}

#define WMMA_F16(A, B, C) \
  __builtin_amdgcn_wmma_f32_16x16x32_f16(false, (A), false, (B), (short)0, (C), false, false)

// ---------------------------------------------------------------------------
// Tensor Data Mover: DMA a 2D f16 tile (tile_d1 rows x tile_d0 elements) from
// global into LDS, inserting pad_amount after every pad_interval so the tile
// lands with the padded SK row stride. D# layout per CDNA5 ISA (§8.3/8.4).
// ---------------------------------------------------------------------------
#if __has_builtin(__builtin_amdgcn_tensor_load_to_lds)
#define HAVE_TDM 1
__device__ __forceinline__ void tdm_load_2d_f16(unsigned int lds_addr, const void* gptr,
                                                unsigned int tensor_d0, unsigned int tensor_d1,
                                                unsigned long long stride0_elems,
                                                unsigned int tile_d0, unsigned int tile_d1,
                                                unsigned int pad_interval_code,
                                                unsigned int pad_amount_code)
{
  const unsigned long long ga = (unsigned long long)gptr;
  u32x4 g0;
  g0[0] = 1u;                                              // count=1, user descriptor
  g0[1] = lds_addr;                                        // LDS byte address
  g0[2] = (unsigned int)(ga & 0xFFFFFFFFu);                // global_addr[31:0]
  g0[3] = (unsigned int)((ga >> 32) & 0x01FFFFFFu) | (2u << 30);  // [56:32] + type=2
  i32x8 g1;
  g1[0] = (int)((1u << 16)                                 // data_size = 1 (2 bytes)
              | (1u << 20)                                 // pad_enable
              | (pad_interval_code << 22)                  // pad interval code
              | (pad_amount_code << 25));                  // pad amount code
  g1[1] = (int)((tensor_d0 & 0xFFFFu) << 16);              // tensor_dim0[15:0] @ bits63:48
  g1[2] = (int)((tensor_d0 >> 16) | ((tensor_d1 & 0xFFFFu) << 16));
  g1[3] = (int)((tensor_d1 >> 16) | (tile_d0 << 16));      // tile_dim0 @ bits127:112
  g1[4] = (int)(tile_d1);                                  // tile_dim1; tile_dim2 = 0
  g1[5] = (int)(stride0_elems & 0xFFFFFFFFu);              // tensor_dim0_stride[31:0]
  g1[6] = (int)((stride0_elems >> 32) & 0xFFFFu);          // stride0[47:32]; stride1=0
  g1[7] = 0;
  const i32x4 gz = {0, 0, 0, 0};
#if defined(__clang_major__) && (__clang_major__ >= 23)
  const i32x8 gz8 = {0, 0, 0, 0, 0, 0, 0, 0};
  __builtin_amdgcn_tensor_load_to_lds(g0, g1, gz, gz, gz8, 0);
#else
  __builtin_amdgcn_tensor_load_to_lds(g0, g1, gz, gz, 0);
#endif
}
#else
#define HAVE_TDM 0
#endif

// ---------------------------------------------------------------------------
// Weight transpose + f16 convert: W[k][n] f32 -> Wt[n][k] f16 (64x64 LDS tiles,
// coalesced both directions). Run once per weight so GEMM B-tiles are pure
// byte-copies for the TDM.
// ---------------------------------------------------------------------------
__global__ __launch_bounds__(256)
void wtrans_kernel(const float* __restrict__ W, _Float16* __restrict__ Wt)
{
  __shared__ _Float16 t[64 * 72];
  const int tid = threadIdx.x;
  const int n0 = blockIdx.x * 64;
  const int k0 = blockIdx.y * 64;
#pragma unroll
  for (int i = 0; i < 16; ++i) {
    int idx = tid + i * 256;
    int k = idx >> 6, n = idx & 63;
    t[n * 72 + k] = (_Float16)W[(size_t)(k0 + k) * D_MODEL + n0 + n];
  }
  __syncthreads();
#pragma unroll
  for (int i = 0; i < 16; ++i) {
    int idx = tid + i * 256;
    int n = idx >> 6, k = idx & 63;
    Wt[(size_t)(n0 + n) * D_MODEL + k0 + k] = t[n * 72 + k];
  }
}

// ---------------------------------------------------------------------------
// GEMM: C[8192,1024] = A(f32) @ W + bias via f16 WMMA, f32 accumulate.
// B tiles DMA'd by the Tensor Data Mover from pre-transposed f16 weights into
// double-buffered LDS (issued at iteration top so the DMA overlaps the whole
// A-staging window); A staged fp32->f16 with batched loads so global_load_b128s
// overlap each other. 8 waves, block tile 128x128x64, 32x64 per wave.
// mode 0/1: f16 [b,h,s,d]; mode 2: f16 [b,h,d,s] (V^T); mode 3: f32 [m,n].
// ---------------------------------------------------------------------------
#define TM 128
#define TN 128
#define TK 64
#define SK (TK + 8)   // padded LDS row stride in halves (144B)

__global__ __launch_bounds__(256)
void gemm_proj_kernel(const float* __restrict__ A, const _Float16* __restrict__ Wt,
                      const float* __restrict__ bias, void* __restrict__ outp, int mode)
{
  __shared__ _Float16 As[TM * SK];
  __shared__ _Float16 Bs[2][TN * SK];
  const int tid  = threadIdx.x;
  const int lane = tid & 31;
  const int wave = tid >> 5;
  const int wm = (wave & 3) * 32;
  const int wn = (wave >> 2) * 64;
  const int tile_m = blockIdx.x * TM;
  const int tile_n = blockIdx.y * TN;

  v8f acc[2][4];
#pragma unroll
  for (int i = 0; i < 2; ++i)
#pragma unroll
    for (int j = 0; j < 4; ++j) acc[i][j] = v8f_zero();

#if HAVE_TDM
  const unsigned int bs_base = (unsigned int)(size_t)(void*)&Bs[0][0];
  const unsigned int bs_size = (unsigned int)(TN * SK * sizeof(_Float16));
  // Prologue: DMA first B tile (128 rows x 64 halves, row stride 1024 elems;
  // pad codes: interval 4 -> 32 DWORDs = 128B (one tile row), amount 3 -> 4
  // DWORDs = 16B, giving the SK=72-half padded stride).
  if (wave == 0)
    tdm_load_2d_f16(bs_base, Wt + (size_t)tile_n * D_MODEL, D_MODEL, D_MODEL,
                    D_MODEL, TK, TN, 4u, 3u);
#endif

  for (int k0 = 0; k0 < D_MODEL; k0 += TK) {
    const int cur = (k0 / TK) & 1;
#if HAVE_TDM
    // Issue next B-tile DMA first: it overlaps the whole A-staging phase.
    if (wave == 0 && (k0 + TK < D_MODEL))
      tdm_load_2d_f16(bs_base + (1 - cur) * bs_size,
                      Wt + (size_t)tile_n * D_MODEL + (k0 + TK), D_MODEL, D_MODEL,
                      D_MODEL, TK, TN, 4u, 3u);
#endif
    // Stage A: batch all 8 global float4 loads so they overlap, then convert.
    float4 tmp[8];
#pragma unroll
    for (int i = 0; i < 8; ++i) {
      int idx = tid + i * 256;
      int m  = idx >> 4;
      int k4 = (idx & 15) << 2;
      tmp[i] = *(const float4*)(A + (size_t)(tile_m + m) * D_MODEL + k0 + k4);
    }
#pragma unroll
    for (int i = 0; i < 8; ++i) {
      int idx = tid + i * 256;
      int m  = idx >> 4;
      int k4 = (idx & 15) << 2;
      _Float16* dst = &As[m * SK + k4];
      dst[0] = (_Float16)tmp[i].x; dst[1] = (_Float16)tmp[i].y;
      dst[2] = (_Float16)tmp[i].z; dst[3] = (_Float16)tmp[i].w;
    }
#if HAVE_TDM
    if (wave == 0) {
      if (k0 + TK < D_MODEL) {
        __builtin_amdgcn_s_wait_tensorcnt(1);   // tile k0 complete, k0+TK in flight
      } else {
        __builtin_amdgcn_s_wait_tensorcnt(0);
      }
    }
#else
    // Fallback: VALU-stage B transposed from Wt.
#pragma unroll
    for (int i = 0; i < 8; ++i) {
      int idx = tid + i * 256;
      int n  = idx >> 4;
      int k4 = (idx & 15) << 2;
      const _Float16* wp = Wt + (size_t)(tile_n + n) * D_MODEL + k0 + k4;
      _Float16* dst = &Bs[cur][n * SK + k4];
      dst[0] = wp[0]; dst[1] = wp[1]; dst[2] = wp[2]; dst[3] = wp[3];
    }
#endif
    __syncthreads();
#pragma unroll
    for (int kk = 0; kk < TK; kk += 32) {
      v16h af[2], bf[4];
#pragma unroll
      for (int i = 0; i < 2; ++i)
        af[i] = load_fragA(&As[(wm + i * 16) * SK + kk], SK, lane);
#pragma unroll
      for (int j = 0; j < 4; ++j)
        bf[j] = load_fragB(&Bs[cur][(wn + j * 16) * SK + kk], SK, lane);
#pragma unroll
      for (int i = 0; i < 2; ++i)
#pragma unroll
        for (int j = 0; j < 4; ++j)
          acc[i][j] = WMMA_F16(af[i], bf[j], acc[i][j]);
    }
    __syncthreads();
  }

  // Epilogue: bias add + scatter per mode.
  const int g = lane >> 4, cn = lane & 15;
#pragma unroll
  for (int i = 0; i < 2; ++i) {
#pragma unroll
    for (int j = 0; j < 4; ++j) {
      const int n = tile_n + wn + j * 16 + cn;
      const float bv = bias[n];
#pragma unroll
      for (int r = 0; r < 8; ++r) {
        const int m = tile_m + wm + i * 16 + r + 8 * g;
        const float val = acc[i][j][r] + bv;
        if (mode == 3) {
          ((float*)outp)[(size_t)m * D_MODEL + n] = val;
        } else {
          const int b = m >> 11, s = m & (SEQ - 1);
          const int h = n >> 6,  d = n & (HEAD_DIM - 1);
          const size_t bh = (size_t)(b * NUM_HEAD + h);
          size_t dst;
          if (mode == 2) dst = (bh * HEAD_DIM + d) * SEQ + s;   // V^T [bh][d][s]
          else           dst = (bh * SEQ + s) * HEAD_DIM + d;   // Q/K [bh][s][d]
          ((_Float16*)outp)[dst] = (_Float16)val;
        }
      }
    }
  }
}

// ---------------------------------------------------------------------------
// Causal flash attention: 4 independent waves per block, 16 queries per wave.
// 32-key blocks: scores via 4 WMMAs, online softmax (16-lane shuffles), P
// relaid through per-wave LDS, O += P@V via 4 WMMAs. Next-block K/V prefetch.
// ---------------------------------------------------------------------------
__global__ __launch_bounds__(128)
void attn_kernel(const _Float16* __restrict__ Qf, const _Float16* __restrict__ Kf,
                 const _Float16* __restrict__ Vt, float* __restrict__ Obuf)
{
  constexpr int PS = 40;
  __shared__ _Float16 pbuf[4 * 16 * PS];
  const int tid  = threadIdx.x;
  const int lane = tid & 31;
  const int wave = tid >> 5;
  const int bh = blockIdx.y;
  const int b = bh >> 4, h = bh & 15;
  const int q0 = blockIdx.x * 64 + wave * 16;
  const int g = lane >> 4, cn = lane & 15;
  const float LOG2E = 1.44269504088896340736f;
  const float scale = 0.125f;

  const _Float16* Qbase = Qf + ((size_t)bh * SEQ + q0) * HEAD_DIM;
  const v16h aq0 = load_fragA(Qbase, HEAD_DIM, lane);
  const v16h aq1 = load_fragA(Qbase + 32, HEAD_DIM, lane);

  v8f o[4];
  float mrow[8], lrow[8];
#pragma unroll
  for (int dt = 0; dt < 4; ++dt) o[dt] = v8f_zero();
#pragma unroll
  for (int r = 0; r < 8; ++r) { mrow[r] = -__builtin_inff(); lrow[r] = 0.0f; }

  _Float16* pw = &pbuf[wave * 16 * PS];
  const int kmax = q0 + 15;
  for (int kb = 0; kb <= kmax; kb += 32) {
    if (kb + 32 <= kmax) {   // prefetch next key block (K rows + V^T rows)
      __builtin_prefetch(Kf + ((size_t)bh * SEQ + kb + 32) * HEAD_DIM + lane * 64, 0, 1);
      __builtin_prefetch(Vt + ((size_t)bh * HEAD_DIM + (lane & 31)) * SEQ + kb + 32, 0, 1);
    }
    float sv[2][8];
#pragma unroll
    for (int t = 0; t < 2; ++t) {
      const _Float16* Kbase = Kf + ((size_t)bh * SEQ + kb + t * 16) * HEAD_DIM;
      const v16h bk0 = load_fragB(Kbase, HEAD_DIM, lane);
      const v16h bk1 = load_fragB(Kbase + 32, HEAD_DIM, lane);
      v8f s = v8f_zero();
      s = WMMA_F16(aq0, bk0, s);
      s = WMMA_F16(aq1, bk1, s);
      const int j = kb + t * 16 + cn;
#pragma unroll
      for (int r = 0; r < 8; ++r) {
        const int q = q0 + r + 8 * g;
        sv[t][r] = (j <= q) ? s[r] * scale : -__builtin_inff();
      }
    }
    float alpha[8];
#pragma unroll
    for (int r = 0; r < 8; ++r) {
      float mx = fmaxf(sv[0][r], sv[1][r]);
#pragma unroll
      for (int off = 1; off < 16; off <<= 1) mx = fmaxf(mx, __shfl_xor(mx, off, 32));
      const float nm = fmaxf(mrow[r], mx);
      const float al = __builtin_amdgcn_exp2f((mrow[r] - nm) * LOG2E);
      const float p0 = __builtin_amdgcn_exp2f((sv[0][r] - nm) * LOG2E);
      const float p1 = __builtin_amdgcn_exp2f((sv[1][r] - nm) * LOG2E);
      float ps = p0 + p1;
#pragma unroll
      for (int off = 1; off < 16; off <<= 1) ps += __shfl_xor(ps, off, 32);
      lrow[r] = lrow[r] * al + ps;
      mrow[r] = nm;
      alpha[r] = al;
      const int row = r + 8 * g;
      pw[row * PS + cn]      = (_Float16)p0;
      pw[row * PS + 16 + cn] = (_Float16)p1;
    }
#pragma unroll
    for (int dt = 0; dt < 4; ++dt)
#pragma unroll
      for (int r = 0; r < 8; ++r) o[dt][r] *= alpha[r];
    const v16h pf = load_fragA(pw, PS, lane);
#pragma unroll
    for (int dt = 0; dt < 4; ++dt) {
      const _Float16* Vbase = Vt + ((size_t)bh * HEAD_DIM + dt * 16) * SEQ + kb;
      const v16h bv = load_fragB(Vbase, SEQ, lane);
      o[dt] = WMMA_F16(pf, bv, o[dt]);
    }
  }

#pragma unroll
  for (int r = 0; r < 8; ++r) {
    const float inv = 1.0f / lrow[r];
    const int q = q0 + r + 8 * g;
#pragma unroll
    for (int dt = 0; dt < 4; ++dt) {
      const int col = h * HEAD_DIM + dt * 16 + cn;
      Obuf[((size_t)b * SEQ + q) * D_MODEL + col] = o[dt][r] * inv;
    }
  }
}

// ---------------------------------------------------------------------------
extern "C" void kernel_launch(void* const* d_in, const int* in_sizes, int n_in,
                              void* d_out, int out_size, void* d_ws, size_t ws_size,
                              hipStream_t stream) {
  (void)in_sizes; (void)n_in; (void)out_size; (void)ws_size;
  const float* q_in = (const float*)d_in[0];
  const float* k_in = (const float*)d_in[1];
  const float* v_in = (const float*)d_in[2];
  const float* Wq = (const float*)d_in[3];
  const float* bq = (const float*)d_in[4];
  const float* Wk = (const float*)d_in[5];
  const float* bk = (const float*)d_in[6];
  const float* Wv = (const float*)d_in[7];
  const float* bv = (const float*)d_in[8];
  const float* Wo = (const float*)d_in[9];
  const float* bo = (const float*)d_in[10];

  // Workspace: Q(f16) | K(f16) | V^T(f16) | attnO(f32) | 4x W^T(f16)  (~88 MB)
  const size_t elems = (size_t)MROWS * D_MODEL;       // 8Mi
  const size_t welems = (size_t)D_MODEL * D_MODEL;    // 1Mi
  _Float16* Qf = (_Float16*)d_ws;
  _Float16* Kf = Qf + elems;
  _Float16* Vt = Kf + elems;
  float*  Obuf = (float*)(Vt + elems);
  _Float16* Wtq = (_Float16*)(Obuf + elems);
  _Float16* Wtk = Wtq + welems;
  _Float16* Wtv = Wtk + welems;
  _Float16* Wto = Wtv + welems;

  dim3 tgrid(D_MODEL / 64, D_MODEL / 64);             // 16 x 16
  wtrans_kernel<<<tgrid, 256, 0, stream>>>(Wq, Wtq);
  wtrans_kernel<<<tgrid, 256, 0, stream>>>(Wk, Wtk);
  wtrans_kernel<<<tgrid, 256, 0, stream>>>(Wv, Wtv);
  wtrans_kernel<<<tgrid, 256, 0, stream>>>(Wo, Wto);

  dim3 ggrid(MROWS / TM, D_MODEL / TN);               // 64 x 8
  gemm_proj_kernel<<<ggrid, 256, 0, stream>>>(q_in, Wtq, bq, Qf, 0);
  gemm_proj_kernel<<<ggrid, 256, 0, stream>>>(k_in, Wtk, bk, Kf, 1);
  gemm_proj_kernel<<<ggrid, 256, 0, stream>>>(v_in, Wtv, bv, Vt, 2);
  attn_kernel<<<dim3(SEQ / 64, BATCH * NUM_HEAD), 128, 0, stream>>>(Qf, Kf, Vt, Obuf);
  gemm_proj_kernel<<<ggrid, 256, 0, stream>>>(Obuf, Wto, bo, d_out, 3);
}
